// SlidingSelfAttention_27504970563751
// MI455X (gfx1250) — matrix operand order repaired and checked by reading, not compiled, not run on hardware
//
#include <hip/hip_runtime.h>

// Flash-style sliding-window attention for MI455X (gfx1250, wave32, WMMA).
// B=4, S=4096, dm=192 (d=64), window = [i-256, i], fp32 in/out.
//
// Block = 128 queries (8 waves x 16). K (384x64) and V^T (64x384) are
// converted to f16 once per block into WGP LDS (~103 KB of the 320 KB CDNA5
// LDS); the chunk loop runs entirely out of LDS (ds_load_b128 -> v_wmma).
//
// Scores computed TRANSPOSED (St = K·Q^T): each lane owns one query, softmax
// state is per-lane scalar (1 ds_swizzle SWAPX16 per reduction), post-softmax
// P lands in the A-fragment pattern and the V^T·P B-fragment needs only a
// lane^16 half-swap. Output accumulated as O^T.
//
// Only the first chunk can hit the window lower edge and only the last chunk
// the causal edge; online softmax is order-independent, so the <=2 edge
// chunks run first through a masked body (per-lane bitmask select), then the
// interior chunks run with NO mask code. Wave-uniform loop bounds are forced
// into SGPRs via readfirstlane so all chunk control flow is scalar branches
// (EXEC stays all-ones for WMMA by construction).

typedef __attribute__((ext_vector_type(16))) _Float16 v16h;
typedef __attribute__((ext_vector_type(8)))  _Float16 v8h;
typedef __attribute__((ext_vector_type(8)))  float    v8f;
typedef __attribute__((ext_vector_type(4)))  float    v4f;
typedef __attribute__((ext_vector_type(4)))  int      v4i;

#define SEQ    4096
#define DMODEL 192
#define DHEAD  64
#define WIN    256
#define NEGF   (-1e30f)

#define KROWS  384          // staged key rows per block
#define KSTR   72           // K LDS row stride in halves (144B, 16B multiple)
#define VSTR   392          // V^T LDS row stride in halves (784B, 16B multiple)
#define LDS_HALVES (KROWS * KSTR + DHEAD * VSTR)

// SWAPX16: and_mask=0x1F, or_mask=0, xor_mask=0x10 -> lane i <-> lane i^16
__device__ __forceinline__ float swapx16f(float x) {
  return __int_as_float(__builtin_amdgcn_ds_swizzle(__float_as_int(x), 0x401f));
}
__device__ __forceinline__ int swapx16i(int x) {
  return __builtin_amdgcn_ds_swizzle(x, 0x401f);
}

__global__ __launch_bounds__(256)
void slattn_fwd_kernel(const float* __restrict__ qkv, float* __restrict__ out)
{
  extern __shared__ __align__(16) _Float16 smem[];
  _Float16* Kl  = smem;                  // [KROWS][KSTR]  f16 K rows
  _Float16* Vtl = smem + KROWS * KSTR;   // [DHEAD][VSTR]  f16 V^T rows

  const int tid  = threadIdx.x;
  const int lane = tid & 31;
  const int h    = lane >> 4;     // lane half
  const int c    = lane & 15;     // query column owned by this lane
  // wave id is wave-uniform: force it into an SGPR so chunk bounds/branches
  // are scalar (s_cbranch_scc), not exec-masked vector control flow.
  const int w    = __builtin_amdgcn_readfirstlane(tid >> 5);

  const int b     = blockIdx.x >> 5;     // 4 batches
  const int qblk  = blockIdx.x & 31;     // 32 query-blocks of 128 per batch
  const int Q0    = qblk * 128;
  const int sbase = Q0 - 256;            // LDS row 0 == key 'sbase'
  const int qt0   = Q0 + w * 16;         // this wave's 16-query tile base (SGPR)
  const int iq    = qt0 + c;             // this lane's query row

  const float* base = qkv + (size_t)b * SEQ * DMODEL;
  const float SCL = 0.125f * 1.44269504088896340736f; // 1/sqrt(64) * log2(e)

  // ================= cooperative staging: K and V^T -> LDS (f16) ==========
  for (int tsk = tid; tsk < 2 * KROWS; tsk += 256) {
    const int j  = tsk >> 1;
    const int d0 = (tsk & 1) * 32;
    int key = sbase + j;
    key = key < 0 ? 0 : key;             // clamped rows are never read
    const float* kr = base + (size_t)key * DMODEL + 64 + d0;
    _Float16* dst = Kl + j * KSTR + d0;
    v4f f[8];
    #pragma unroll
    for (int u = 0; u < 8; ++u) f[u] = *(const v4f*)(kr + 4 * u);
    #pragma unroll
    for (int u = 0; u < 4; ++u) {
      v8h pk;
      #pragma unroll
      for (int t = 0; t < 4; ++t) {
        pk[t]     = (_Float16)f[2 * u][t];
        pk[4 + t] = (_Float16)f[2 * u + 1][t];
      }
      *(v8h*)(dst + 8 * u) = pk;
    }
  }
  for (int tsk = tid; tsk < 48 * 8; tsk += 256) {
    const int dg = tsk & 7;
    const int ko = tsk >> 3;
    float f[8][8];
    #pragma unroll
    for (int j2 = 0; j2 < 8; ++j2) {
      int key = sbase + 8 * ko + j2;
      key = key < 0 ? 0 : key;
      const float* vr = base + (size_t)key * DMODEL + 128 + 8 * dg;
      v4f a  = *(const v4f*)(vr);
      v4f bb = *(const v4f*)(vr + 4);
      #pragma unroll
      for (int t = 0; t < 4; ++t) { f[j2][t] = a[t]; f[j2][4 + t] = bb[t]; }
    }
    #pragma unroll
    for (int dd = 0; dd < 8; ++dd) {
      v8h pk;
      #pragma unroll
      for (int j2 = 0; j2 < 8; ++j2) pk[j2] = (_Float16)f[j2][dd];
      *(v8h*)(Vtl + (8 * dg + dd) * VSTR + 8 * ko) = pk;
    }
  }
  __syncthreads();

  // ================= per-wave flash attention over LDS ====================
  v16h bq0, bq1;   // Q^T B-fragments (col = q = c, K = d = kb + 8h + e (+8 for e>=8))
  {
    const float* qrow = base + (size_t)iq * DMODEL;
    #pragma unroll
    for (int e = 0; e < 16; ++e) {
      bq0[e] = (_Float16)(qrow[     8 * h + e + (e >= 8 ? 8 : 0)] * SCL);
      bq1[e] = (_Float16)(qrow[32 + 8 * h + e + (e >= 8 ? 8 : 0)] * SCL);
    }
  }

  v8f acc[4];      // O^T accumulators: row = d = 16nd + r + 8h, col = q = c
  #pragma unroll
  for (int nd = 0; nd < 4; ++nd)
    #pragma unroll
    for (int r = 0; r < 8; ++r) acc[nd][r] = 0.0f;

  float m = NEGF, lsum = 0.0f;   // per-lane (per-query) softmax state

  const int kc_lo   = (qt0 >= WIN ? (qt0 - WIN) : 0) & ~31;   // SGPR
  const int kc_last = (qt0 + 15) & ~31;                       // SGPR

  // per-lane LDS base offsets (halves)
  const _Float16* KlLane  = Kl  + c * KSTR + 8 * h;
  const _Float16* VtlLane = Vtl + c * VSTR + 16 * h;

  // ---- one chunk of 32 keys; MASKED folds away at the unmasked call site
  auto processChunk = [&](const int kc, const bool MASKED) {
    const int jb = kc - sbase;   // LDS row of this chunk's first key (SGPR)

    // St = K · Q^T : 2 key-subtiles x 2 d-chunks, A-frags from LDS.
    v8f st[2];
    #pragma unroll
    for (int n = 0; n < 2; ++n) {
      const _Float16* kr = KlLane + (size_t)(jb + 16 * n) * KSTR;
      v8h a0 = *(const v8h*)(kr);
      v8h a1 = *(const v8h*)(kr + 16);
      v8h a2 = *(const v8h*)(kr + 32);
      v8h a3 = *(const v8h*)(kr + 48);
      v16h ak0, ak1;
      #pragma unroll
      for (int e = 0; e < 8; ++e) {
        ak0[e] = a0[e]; ak0[e + 8] = a1[e];
        ak1[e] = a2[e]; ak1[e + 8] = a3[e];
      }
      v8f cc;
      #pragma unroll
      for (int r = 0; r < 8; ++r) cc[r] = 0.0f;
      cc = __builtin_amdgcn_wmma_f32_16x16x32_f16(false, ak0, false, bq0, (short)0, cc, false, false);
      cc = __builtin_amdgcn_wmma_f32_16x16x32_f16(false, ak1, false, bq1, (short)0, cc, false, false);
      st[n] = cc;
    }

    // Edge mask: per-lane validity bitmask over kappa=0..31. Pre-shift by 8h
    // so every per-element shift amount is a compile-time constant.
    unsigned Ma = 0xffffffffu;
    if (MASKED) {
      const int lo2 = iq - WIN - kc;   // valid kappa >= lo2
      const int hi2 = iq - kc;         // valid kappa <= hi2
      const unsigned Ml = (lo2 > 0) ? (0xffffffffu << (lo2 & 31)) : 0xffffffffu;
      const unsigned Mh = (hi2 >= 31) ? 0xffffffffu
                         : ((hi2 < 0) ? 0u : (0xffffffffu >> ((31 - hi2) & 31)));
      Ma = (Ml & Mh) >> (8 * h);       // bit r   -> kappa = r + 8h (n=0)
    }                                  // bit 16+r-> kappa = 16 + r + 8h (n=1)

    float t0[8], t1[8];
    #pragma unroll
    for (int r = 0; r < 8; ++r) {
      float a  = st[0][r];
      float bb = st[1][r];
      if (MASKED) {
        const int sel0 = ((int)(Ma << (31 - r))) >> 31;   // 0 or -1
        const int sel1 = ((int)(Ma << (15 - r))) >> 31;
        a  = __int_as_float((sel0 & __float_as_int(a))  | (~sel0 & __float_as_int(NEGF)));
        bb = __int_as_float((sel1 & __float_as_int(bb)) | (~sel1 & __float_as_int(NEGF)));
      }
      t0[r] = a; t1[r] = bb;
    }

    // online softmax: partner lane (lane^16) holds the other key half.
    float mx = fmaxf(t0[0], t1[0]);
    #pragma unroll
    for (int r = 1; r < 8; ++r) mx = fmaxf(mx, fmaxf(t0[r], t1[r]));
    mx = fmaxf(mx, swapx16f(mx));

    const float mn   = fmaxf(m, mx);
    const float corr = exp2f(m - mn);
    m = mn;

    float p0[8], p1[8], rs = 0.0f;
    #pragma unroll
    for (int r = 0; r < 8; ++r) {
      p0[r] = exp2f(t0[r] - mn);
      p1[r] = exp2f(t1[r] - mn);
      rs += p0[r] + p1[r];
    }
    rs += swapx16f(rs);
    lsum = lsum * corr + rs;

    #pragma unroll
    for (int nd = 0; nd < 4; ++nd)
      #pragma unroll
      for (int r = 0; r < 8; ++r) acc[nd][r] *= corr;

    // B-fragment of P (32k x 16q) via lane^16 half-swap.
    v8h h0, h1;
    #pragma unroll
    for (int r = 0; r < 8; ++r) { h0[r] = (_Float16)p0[r]; h1[r] = (_Float16)p1[r]; }
    v4i i0 = __builtin_bit_cast(v4i, h0);
    v4i i1 = __builtin_bit_cast(v4i, h1);
    v4i o0, o1;
    #pragma unroll
    for (int j = 0; j < 4; ++j) { o0[j] = swapx16i(i0[j]); o1[j] = swapx16i(i1[j]); }
    const v4i blo = h ? o1 : i0;   // h=0: own keys 0..7    ; h=1: partner 16..23
    const v4i bhi = h ? i1 : o0;   // h=0: partner keys 8..15; h=1: own 24..31
    const v8h plo = __builtin_bit_cast(v8h, blo);
    const v8h phi = __builtin_bit_cast(v8h, bhi);
    v16h bp;
    #pragma unroll
    for (int e = 0; e < 8; ++e) { bp[e] = plo[e]; bp[e + 8] = phi[e]; }

    // O^T += V^T(16d x 32k) · P(32k x 16q), A-frags from LDS V^T.
    #pragma unroll
    for (int nd = 0; nd < 4; ++nd) {
      const _Float16* vr = VtlLane + (size_t)(16 * nd) * VSTR + jb;
      v8h lo = *(const v8h*)(vr);
      v8h hi = *(const v8h*)(vr + 8);
      v16h av;
      #pragma unroll
      for (int e = 0; e < 8; ++e) { av[e] = lo[e]; av[e + 8] = hi[e]; }
      acc[nd] = __builtin_amdgcn_wmma_f32_16x16x32_f16(false, av, false, bp,
                                                       (short)0, acc[nd], false, false);
    }
  };

  // Edge chunks first (order-independent accumulation), single static body.
  #pragma unroll 1
  for (int e = 0; e < 2; ++e) {
    if (e && kc_last == kc_lo) break;
    processChunk(e ? kc_last : kc_lo, true);
  }
  // Interior chunks: mask-free; pinned to a single static body.
  #pragma unroll 1
  for (int kc = kc_lo + 32; kc < kc_last; kc += 32)
    processChunk(kc, false);

  // ---- epilogue: O^T layout gives each lane 8 contiguous d per subtile.
  const float inv = 1.0f / lsum;
  float* orow = out + ((size_t)b * SEQ + iq) * DHEAD;
  #pragma unroll
  for (int nd = 0; nd < 4; ++nd) {
    v4f lo, hi;
    #pragma unroll
    for (int t = 0; t < 4; ++t) {
      lo[t] = acc[nd][t]     * inv;
      hi[t] = acc[nd][4 + t] * inv;
    }
    *(v4f*)(orow + 16 * nd + 8 * h)     = lo;
    *(v4f*)(orow + 16 * nd + 8 * h + 4) = hi;
  }
}

extern "C" void kernel_launch(void* const* d_in, const int* in_sizes, int n_in,
                              void* d_out, int out_size, void* d_ws, size_t ws_size,
                              hipStream_t stream) {
  const float* qkv = (const float*)d_in[0];
  float* out = (float*)d_out;
  // 4 batches * (4096/128) query-blocks = 128 blocks of 256 threads (8 waves).
  const int blocks = in_sizes[0] / (DMODEL * 128);
  const size_t lds_bytes = (size_t)LDS_HALVES * sizeof(_Float16); // ~103 KB < 320 KB WGP LDS
  slattn_fwd_kernel<<<dim3(blocks), dim3(256), lds_bytes, stream>>>(qkv, out);
}
